// SelfAttentionModule_24008867184968
// MI455X (gfx1250) — matrix operand order repaired and checked by reading, compiled, not run on hardware
//
#include <hip/hip_runtime.h>
#include <hip/hip_bf16.h>

typedef __attribute__((ext_vector_type(16))) _Float16 v16h;
typedef __attribute__((ext_vector_type(8)))  _Float16 v8h;
typedef __attribute__((ext_vector_type(4)))  _Float16 v4h;
typedef __attribute__((ext_vector_type(2)))  _Float16 v2h;
typedef __attribute__((ext_vector_type(8)))  float    v8f;

#define BN_EPS 1e-5f

// ---------------------------------------------------------------- utilities

__global__ void cvt_f32_to_f16(const float* __restrict__ src,
                               _Float16* __restrict__ dst, int n) {
  int i = blockIdx.x * blockDim.x + threadIdx.x;
  int stride = gridDim.x * blockDim.x;
  for (; i < n; i += stride) dst[i] = (_Float16)src[i];
}

// Fold conv bias (+ optional inference BN) into per-channel affine:
//   y = conv * scale + shift ; relu flag per channel.
__global__ void make_coef(const float* __restrict__ bias,
                          const float* __restrict__ bn,   // (4,C) rows g,b,m,v or null
                          float* __restrict__ scale, float* __restrict__ shift,
                          unsigned char* __restrict__ relu, int C, int hasBN) {
  int c = blockIdx.x * blockDim.x + threadIdx.x;
  if (c >= C) return;
  if (hasBN) {
    float g = bn[c], beta = bn[C + c], m = bn[2 * C + c], v = bn[3 * C + c];
    float s = g * rsqrtf(v + BN_EPS);
    scale[c] = s;
    shift[c] = (bias[c] - m) * s + beta;
    relu[c] = 1;
  } else {
    scale[c] = 1.0f;
    shift[c] = bias[c];
    relu[c] = 0;
  }
}

// ------------------------------------------------------------ WMMA 1x1-conv
// C(MxN) = A(MxK,f16 row-major) * B(KxN,f16) ; per-row affine (+relu) epilogue.
// Block = 256 threads = 8 wave32 -> 64(M) x 64(N) tile.
// Wave (m=wave&3, n=wave>>2) computes 16(M) x 32(N): two 16x16 WMMA tiles
// sharing one A fragment. B is staged in LDS *transposed* ([n][k], stride 40)
// so each lane's 16 K-contiguous halves load as two ds_load_b128.
// Double-buffered LDS: one barrier per K-step; next tile's global loads are
// issued before this tile's WMMAs.
// Requires: M%64==0, N%64==0, K%32==0 (true for all launches here).
__global__ void wmma_gemm_affine(const _Float16* __restrict__ A,
                                 const _Float16* __restrict__ B,
                                 _Float16* __restrict__ Ch, float* __restrict__ Cf,
                                 const float* __restrict__ scale,
                                 const float* __restrict__ shift,
                                 const unsigned char* __restrict__ relu,
                                 int M, int N, int K,
                                 unsigned long long strideB,   // per-batch elems
                                 unsigned long long strideC,   // per-batch elems
                                 int outF32) {
  typedef unsigned long long ull;
  __shared__ _Float16 ldsT[2][64 * 40];   // [buf][n(64)][k(32)+pad8]

  const int tid  = threadIdx.x;
  const int lane = tid & 31;
  const int wave = tid >> 5;
  const int mBase = blockIdx.y * 64 + (wave & 3) * 16;
  const int nb0   = blockIdx.x * 64;
  const int nWave = (wave >> 2) * 32;          // 0 or 32

  // A-fragment lane mapping (contiguous k order used here, same as B; 16-bit A 16x32):
  //  lanes 0-15: row=lane,    halves0-15 => K=k0..+15
  //  lanes16-31: row=lane-16, halves0-15 => K=k0+16..+31
  const int rowA = mBase + (lane & 15);
  const int ka   = (lane >> 4) << 4;           // 0 or 16
  // B-fragment: lane col = lane&15 (+16 for 2nd tile); halves h => K=(lane>>4)*16+h
  const int kb   = (lane >> 4) << 4;           // 0 or 16
  const int col0 = nWave + (lane & 15);
  const int col1 = col0 + 16;

  const _Float16* Bb = B + (ull)blockIdx.z * strideB;

  // cooperative transposed fill: thread t handles K-pair kp, 4 N columns
  const int kp = (tid >> 4) * 2;               // 0,2,..,30
  const int n4 = (tid & 15) * 4;               // 0,4,..,60

  v8f acc0 = {}, acc1 = {};

  // prologue: fill buffer 0 with K-rows [0,32)
  {
    const _Float16* b0 = Bb + (ull)kp * N + nb0 + n4;
    const _Float16* b1 = b0 + N;
    v4h r0 = *(const v4h*)b0;
    v4h r1 = *(const v4h*)b1;
#pragma unroll
    for (int j = 0; j < 4; ++j)
      *(v2h*)&ldsT[0][(n4 + j) * 40 + kp] = (v2h){r0[j], r1[j]};
  }

  int buf = 0;
  for (int k0 = 0; k0 < K; k0 += 32) {
    __syncthreads();

    // issue next tile's global loads + LDS fill before this tile's WMMAs
    if (k0 + 32 < K) {
      const _Float16* b0 = Bb + (ull)(k0 + 32 + kp) * N + nb0 + n4;
      const _Float16* b1 = b0 + N;
      v4h r0 = *(const v4h*)b0;
      v4h r1 = *(const v4h*)b1;
#pragma unroll
      for (int j = 0; j < 4; ++j)
        *(v2h*)&ldsT[buf ^ 1][(n4 + j) * 40 + kp] = (v2h){r0[j], r1[j]};
    }
    if (k0 + 64 < K)   // stream the tile after that into cache (global_prefetch)
      __builtin_prefetch(Bb + (ull)(k0 + 64 + kp) * N + nb0 + n4, 0, 1);

    // A fragment: two 16B global loads
    const _Float16* ap = A + (ull)rowA * K + k0;
    v8h alo = *(const v8h*)(ap + ka);
    v8h ahi = *(const v8h*)(ap + 8 + ka);
    v16h af, bf0, bf1;
#pragma unroll
    for (int h = 0; h < 8; ++h) { af[h] = alo[h]; af[8 + h] = ahi[h]; }

    // B fragments: 16 contiguous halves each -> 2x ds_load_b128
    {
      const _Float16* l0 = &ldsT[buf][col0 * 40 + kb];
      const _Float16* l1 = &ldsT[buf][col1 * 40 + kb];
      v8h b0lo = *(const v8h*)l0;
      v8h b0hi = *(const v8h*)(l0 + 8);
      v8h b1lo = *(const v8h*)l1;
      v8h b1hi = *(const v8h*)(l1 + 8);
#pragma unroll
      for (int h = 0; h < 8; ++h) {
        bf0[h] = b0lo[h]; bf0[8 + h] = b0hi[h];
        bf1[h] = b1lo[h]; bf1[8 + h] = b1hi[h];
      }
    }

    acc0 = __builtin_amdgcn_wmma_f32_16x16x32_f16(
        false, af, false, bf0, (short)0, acc0, false, false);
    acc1 = __builtin_amdgcn_wmma_f32_16x16x32_f16(
        false, af, false, bf1, (short)0, acc1, false, false);
    buf ^= 1;
  }

  // C layout: lane col = lane&15 ; VGPR r -> row = r + 8*(lane>>4)
  const ull cb = (ull)blockIdx.z * strideC;
  const int n0 = nb0 + nWave + (lane & 15);
  const int mo = (lane >> 4) << 3;
#pragma unroll
  for (int r = 0; r < 8; ++r) {
    int m = mBase + mo + r;
    float s = scale[m], sh = shift[m];
    float y0 = acc0[r] * s + sh;
    float y1 = acc1[r] * s + sh;
    if (relu[m]) { y0 = fmaxf(y0, 0.0f); y1 = fmaxf(y1, 0.0f); }
    ull idx = cb + (ull)m * (ull)N + n0;
    if (outF32) { Cf[idx] = y0; Cf[idx + 16] = y1; }
    else        { Ch[idx] = (_Float16)y0; Ch[idx + 16] = (_Float16)y1; }
  }
}

// ---------------------------------------------------------- 7x7 local attn
// sim[t] = <key(.,p+dt), query(.,p)> over 49 taps (zero outside -> sim=0,
// still included in softmax, matching zero-padded unfold); then
// ctx = sum_t softmax(sim)[t] * value(.,p+dt).
__global__ void local_attn(const _Float16* __restrict__ key,
                           const _Float16* __restrict__ query,
                           const _Float16* __restrict__ value,
                           _Float16* __restrict__ ctx,
                           int H, int W,
                           unsigned long long kqStride,
                           unsigned long long vStride,
                           unsigned long long cStride) {
  const int HWp = H * W;
  int p = blockIdx.x * blockDim.x + threadIdx.x;
  if (p >= HWp) return;
  const int y = p / W, x = p % W;
  const unsigned long long kq = (unsigned long long)blockIdx.y * kqStride;
  const unsigned long long vb = (unsigned long long)blockIdx.y * vStride;
  const unsigned long long cb = (unsigned long long)blockIdx.y * cStride;

  float qv[64];
#pragma unroll
  for (int c = 0; c < 64; ++c)
    qv[c] = (float)query[kq + (unsigned long long)c * HWp + p];

  float sim[49];
  float mx = 0.0f;                 // padding taps give sim=0, included in max
#pragma unroll
  for (int t = 0; t < 49; ++t) {
    int yy = y + t / 7 - 3, xx = x + t % 7 - 3;
    float s = 0.0f;
    if (yy >= 0 && yy < H && xx >= 0 && xx < W) {
      int p2 = yy * W + xx;
#pragma unroll
      for (int c = 0; c < 64; ++c)
        s += (float)key[kq + (unsigned long long)c * HWp + p2] * qv[c];
    }
    sim[t] = s;
    mx = fmaxf(mx, s);
  }
  float denom = 0.0f;
#pragma unroll
  for (int t = 0; t < 49; ++t) { sim[t] = __expf(sim[t] - mx); denom += sim[t]; }
  const float inv = 1.0f / denom;

#pragma unroll 1
  for (int cv = 0; cv < 128; ++cv) {
    const _Float16* vp = value + vb + (unsigned long long)cv * HWp;
    float a = 0.0f;
#pragma unroll
    for (int t = 0; t < 49; ++t) {
      int yy = y + t / 7 - 3, xx = x + t % 7 - 3;
      if (yy >= 0 && yy < H && xx >= 0 && xx < W)
        a += sim[t] * (float)vp[yy * W + xx];
    }
    ctx[cb + (unsigned long long)cv * HWp + p] = (_Float16)(a * inv);
  }
}

// ------------------------------------------------------------------- launch

extern "C" void kernel_launch(void* const* d_in, const int* in_sizes, int n_in,
                              void* d_out, int out_size, void* d_ws, size_t ws_size,
                              hipStream_t stream) {
  (void)in_sizes; (void)n_in; (void)out_size; (void)ws_size;

  const float* x      = (const float*)d_in[0];
  const float* fk_w1  = (const float*)d_in[1];
  const float* fk_b1  = (const float*)d_in[2];
  const float* fk_bn1 = (const float*)d_in[3];
  const float* fk_w2  = (const float*)d_in[4];
  const float* fk_b2  = (const float*)d_in[5];
  const float* fk_bn2 = (const float*)d_in[6];
  const float* fq_w1  = (const float*)d_in[7];
  const float* fq_b1  = (const float*)d_in[8];
  const float* fq_bn1 = (const float*)d_in[9];
  const float* fq_w2  = (const float*)d_in[10];
  const float* fq_b2  = (const float*)d_in[11];
  const float* fq_bn2 = (const float*)d_in[12];
  const float* fv_w   = (const float*)d_in[13];
  const float* fv_b   = (const float*)d_in[14];
  const float* W_w    = (const float*)d_in[15];
  const float* W_b    = (const float*)d_in[16];
  float* out = (float*)d_out;

  const int Bn = 2, Cin = 256, Ck = 64, Cv = 128, Co = 256, H = 96, Wd = 96;
  const int HW = H * Wd;                       // 9216, divisible by 64
  typedef unsigned long long ull;

  // ---- carve workspace (256B-aligned chunks)
  char* wp = (char*)d_ws;
  auto carve = [&](size_t bytes) -> char* {
    char* p = wp;
    wp += (bytes + 255) & ~(size_t)255;
    return p;
  };
  _Float16* xh    = (_Float16*)carve(sizeof(_Float16) * (size_t)Bn * Cin * HW);
  _Float16* qkvh  = (_Float16*)carve(sizeof(_Float16) * (size_t)Bn * 256 * HW);
  _Float16* keyh  = (_Float16*)carve(sizeof(_Float16) * (size_t)Bn * Ck * HW);
  _Float16* qh    = (_Float16*)carve(sizeof(_Float16) * (size_t)Bn * Ck * HW);
  _Float16* ctxh  = (_Float16*)carve(sizeof(_Float16) * (size_t)Bn * Cv * HW);
  _Float16* Wqkvh = (_Float16*)carve(sizeof(_Float16) * 256 * 256);
  _Float16* Wk2h  = (_Float16*)carve(sizeof(_Float16) * 64 * 64);
  _Float16* Wq2h  = (_Float16*)carve(sizeof(_Float16) * 64 * 64);
  _Float16* Wwh   = (_Float16*)carve(sizeof(_Float16) * 256 * 128);
  float* scQKV = (float*)carve(4 * 256);
  float* shQKV = (float*)carve(4 * 256);
  unsigned char* rlQKV = (unsigned char*)carve(256);
  float* scK2 = (float*)carve(4 * 64);
  float* shK2 = (float*)carve(4 * 64);
  unsigned char* rlK2 = (unsigned char*)carve(64);
  float* scQ2 = (float*)carve(4 * 64);
  float* shQ2 = (float*)carve(4 * 64);
  unsigned char* rlQ2 = (unsigned char*)carve(64);
  float* scW = (float*)carve(4 * 256);
  float* shW = (float*)carve(4 * 256);
  unsigned char* rlW = (unsigned char*)carve(256);

  // ---- f32 -> f16 conversions (fused QKV A-matrix: rows 0..127=value,
  //      128..191=key1, 192..255=query1)
  auto cvt = [&](const float* s, _Float16* d, int n) {
    int blocks = (n + 255) / 256;
    hipLaunchKernelGGL(cvt_f32_to_f16, dim3(blocks), dim3(256), 0, stream, s, d, n);
  };
  cvt(x, xh, Bn * Cin * HW);
  cvt(fv_w,  Wqkvh,             Cv * Cin);
  cvt(fk_w1, Wqkvh + 128 * Cin, Ck * Cin);
  cvt(fq_w1, Wqkvh + 192 * Cin, Ck * Cin);
  cvt(fk_w2, Wk2h, Ck * Ck);
  cvt(fq_w2, Wq2h, Ck * Ck);
  cvt(W_w,   Wwh,  Co * Cv);

  // ---- fold bias/BN into per-channel affine
  auto coef = [&](const float* b, const float* bn, float* sc, float* sh,
                  unsigned char* rl, int C, int hasBN) {
    hipLaunchKernelGGL(make_coef, dim3((C + 63) / 64), dim3(64), 0, stream,
                       b, bn, sc, sh, rl, C, hasBN);
  };
  coef(fv_b,  nullptr, scQKV,       shQKV,       rlQKV,       128, 0);
  coef(fk_b1, fk_bn1,  scQKV + 128, shQKV + 128, rlQKV + 128, 64, 1);
  coef(fq_b1, fq_bn1,  scQKV + 192, shQKV + 192, rlQKV + 192, 64, 1);
  coef(fk_b2, fk_bn2,  scK2, shK2, rlK2, 64, 1);
  coef(fq_b2, fq_bn2,  scQ2, shQ2, rlQ2, 64, 1);
  coef(W_b,   nullptr, scW,  shW,  rlW,  256, 0);

  // ---- fused value/key1/query1 projection: (256x256) x (256xHW) per batch
  hipLaunchKernelGGL(wmma_gemm_affine, dim3(HW / 64, 256 / 64, Bn), dim3(256), 0, stream,
                     Wqkvh, xh, qkvh, (float*)nullptr, scQKV, shQKV, rlQKV,
                     256, HW, Cin, (ull)Cin * HW, (ull)256 * HW, 0);

  // ---- key = BN.ReLU(Wk2 * k1) ; k1 lives at qkvh channels 128..191
  hipLaunchKernelGGL(wmma_gemm_affine, dim3(HW / 64, 1, Bn), dim3(256), 0, stream,
                     Wk2h, qkvh + (size_t)128 * HW, keyh, (float*)nullptr,
                     scK2, shK2, rlK2, Ck, HW, Ck, (ull)256 * HW, (ull)Ck * HW, 0);

  // ---- query = BN.ReLU(Wq2 * q1) ; q1 at qkvh channels 192..255
  hipLaunchKernelGGL(wmma_gemm_affine, dim3(HW / 64, 1, Bn), dim3(256), 0, stream,
                     Wq2h, qkvh + (size_t)192 * HW, qh, (float*)nullptr,
                     scQ2, shQ2, rlQ2, Ck, HW, Ck, (ull)256 * HW, (ull)Ck * HW, 0);

  // ---- 7x7 window attention ; value = qkvh channels 0..127
  hipLaunchKernelGGL(local_attn, dim3((HW + 255) / 256, Bn), dim3(256), 0, stream,
                     keyh, qh, qkvh, ctxh, H, Wd,
                     (ull)Ck * HW, (ull)256 * HW, (ull)Cv * HW);

  // ---- final 1x1 conv: (256x128) x (128xHW) -> fp32 d_out (b,256,H,W)
  hipLaunchKernelGGL(wmma_gemm_affine, dim3(HW / 64, 256 / 64, Bn), dim3(256), 0, stream,
                     Wwh, ctxh, (_Float16*)nullptr, out, scW, shW, rlW,
                     Co, HW, Cv, (ull)Cv * HW, (ull)Co * HW, 1);
}
